// LongTermAttention_75342316306711
// MI455X (gfx1250) — matrix operand order repaired and checked by reading, not compiled, not run on hardware
//
#include <hip/hip_runtime.h>
#include <hip/hip_bf16.h>

// ---------------------------------------------------------------- types
typedef __attribute__((ext_vector_type(16))) __bf16 v16bf;
typedef __attribute__((ext_vector_type(8)))  float  v8f;
typedef __attribute__((ext_vector_type(4)))  unsigned int v4u;

#define E_DIM  1024
#define L_DIM  2048
#define NB_DIM 512
#define Q_DIM  512
#define B_DIM  8
#define H_DIM  16
#define D_DIM  64

// ---------------------------------------------------------------- helpers
__device__ __forceinline__ unsigned short f2bf(float x) {
  unsigned int u = __float_as_uint(x);
  unsigned int r = u + 0x7FFFu + ((u >> 16) & 1u);   // round-to-nearest-even
  return (unsigned short)(r >> 16);
}

// A-matrix 16x32 bf16 fragment: lane holds row (lane&15);
// halves 0..7 = K at (lane>>4)*8, halves 8..15 = K at 16+(lane>>4)*8
__device__ __forceinline__ v16bf load_frag_a(const unsigned short* base,
                                             long stride, int lane) {
  const unsigned short* p = base + (long)(lane & 15) * stride + ((lane >> 4) * 8);
  union { v4u u[2]; v16bf v; } f;
  f.u[0] = *(const v4u*)(p);
  f.u[1] = *(const v4u*)(p + 16);
  return f.v;
}

// B-matrix 32x16 bf16 fragment (stored as [N][K] row-major): lane holds
// column (lane&15); halves 0..7 = K at (lane>>4)*8, halves 8..15 = K at 16+(lane>>4)*8
__device__ __forceinline__ v16bf load_frag_b(const unsigned short* base,
                                             long stride, int lane) {
  const unsigned short* p = base + (long)(lane & 15) * stride + ((lane >> 4) * 8);
  union { v4u u[2]; v16bf v; } f;
  f.u[0] = *(const v4u*)(p);
  f.u[1] = *(const v4u*)(p + 16);
  return f.v;
}

__device__ __forceinline__ v8f wmma_bf16(v16bf a, v16bf b, v8f c) {
  return __builtin_amdgcn_wmma_f32_16x16x32_bf16(
      false, a, false, b, (short)0, c, false, false);
}

// async global -> LDS copy of 16 bytes (GV mode, ASYNCcnt-tracked)
__device__ __forceinline__ void async_ld16(const unsigned short* gptr,
                                           const unsigned short* lptr) {
  unsigned lds = (unsigned)(size_t)lptr;   // low 32 bits = LDS byte offset
  asm volatile("global_load_async_to_lds_b128 %0, %1, off"
               :: "v"(lds), "v"(gptr) : "memory");
}

// ---------------------------------------------------------------- converts
__global__ void cvt_f32_bf16(const float* __restrict__ src,
                             unsigned short* __restrict__ dst, int n) {
  int i = blockIdx.x * 256 + threadIdx.x;
  if (i < n) dst[i] = f2bf(src[i]);
}

// dst[c][r] (bf16) = src[r][c] (f32), tiled 32x32; per-batch strides
__global__ __launch_bounds__(256)
void transpose_f32_bf16(const float* __restrict__ src, long srcBatch, int srcStride,
                        unsigned short* __restrict__ dst, long dstBatch, int dstStride) {
  __shared__ float tile[32][33];
  const float* s = src + (long)blockIdx.z * srcBatch;
  unsigned short* d = dst + (long)blockIdx.z * dstBatch;
  int r0 = blockIdx.y * 32, c0 = blockIdx.x * 32;
  int tx = threadIdx.x & 31, ty = threadIdx.x >> 5;      // 32 x 8
  #pragma unroll
  for (int i = 0; i < 32; i += 8)
    tile[ty + i][tx] = s[(long)(r0 + ty + i) * srcStride + c0 + tx];
  __syncthreads();
  #pragma unroll
  for (int i = 0; i < 32; i += 8)
    d[(long)(c0 + ty + i) * dstStride + r0 + tx] = f2bf(tile[tx][ty + i]);
}

// ---------------------------------------------------------------- generic GEMM
// C[M,N] = A[M,K] * Bt[N,K]^T ; A,Bt bf16 row-major, C f32 or bf16.
// Block tile 128x128, 8 waves (4m x 2n), wave tile 32x64.
// K-step 64, double-buffered LDS filled with async global->LDS loads.
#define KT   64
#define LDSK 72   // padded row stride (halves); 144B = 9*16B, keeps b128 alignment
template <bool CBF16>
__global__ __launch_bounds__(256)
void gemm_bf16(const unsigned short* __restrict__ A, long aBatch, int lda,
               const unsigned short* __restrict__ Bt, long bBatch, int ldb,
               void* __restrict__ Cv, long cBatch, int ldc, int K) {
  __shared__ unsigned short aT[2][128 * LDSK];
  __shared__ unsigned short bT[2][128 * LDSK];
  const int bz = blockIdx.z;
  A  += (long)bz * aBatch;
  Bt += (long)bz * bBatch;
  const int m0 = blockIdx.y * 128, n0 = blockIdx.x * 128;
  const int tid = threadIdx.x, lane = tid & 31, wid = tid >> 5;
  const int wm = wid >> 1, wn = wid & 1;                 // wave tile origin

  v8f acc[2][4];
  #pragma unroll
  for (int i = 0; i < 2; ++i)
    #pragma unroll
    for (int j = 0; j < 4; ++j) acc[i][j] = (v8f){};

  // stage one 128x64 A tile + 128x64 B tile: 2048 16B chunks / 256 thr = 8 each
  auto stage = [&](int k0, unsigned short* aDst, unsigned short* bDst) {
    #pragma unroll
    for (int s = 0; s < 4; ++s) {
      int c = tid + s * 256;
      int row = c >> 3, off = (c & 7) * 8;
      async_ld16(&A[(long)(m0 + row) * lda + k0 + off], &aDst[row * LDSK + off]);
      async_ld16(&Bt[(long)(n0 + row) * ldb + k0 + off], &bDst[row * LDSK + off]);
    }
  };

  const int nk = K / KT;
  stage(0, aT[0], bT[0]);
  for (int it = 0; it < nk; ++it) {
    const int cur = it & 1;
    if (it + 1 < nk) {
      stage((it + 1) * KT, aT[cur ^ 1], bT[cur ^ 1]);
      asm volatile("s_wait_asynccnt 8" ::: "memory");   // 8 newest still in flight
    } else {
      asm volatile("s_wait_asynccnt 0" ::: "memory");
    }
    __syncthreads();
    #pragma unroll
    for (int kk = 0; kk < KT; kk += 32) {
      v16bf af[2], bfr[4];
      #pragma unroll
      for (int i = 0; i < 2; ++i)
        af[i] = load_frag_a(&aT[cur][(wm * 32 + i * 16) * LDSK + kk], LDSK, lane);
      #pragma unroll
      for (int j = 0; j < 4; ++j)
        bfr[j] = load_frag_b(&bT[cur][(wn * 64 + j * 16) * LDSK + kk], LDSK, lane);
      #pragma unroll
      for (int i = 0; i < 2; ++i)
        #pragma unroll
        for (int j = 0; j < 4; ++j)
          acc[i][j] = wmma_bf16(af[i], bfr[j], acc[i][j]);
    }
    __syncthreads();   // all waves done reading before buffer reuse
  }

  // C layout: lane l, vgpr r -> m = r + 8*(l>>4), n = l&15
  const int nl = lane & 15, rowHalf = (lane >> 4) * 8;
  const long cOff = (long)bz * cBatch;
  #pragma unroll
  for (int i = 0; i < 2; ++i)
    #pragma unroll
    for (int j = 0; j < 4; ++j) {
      int mBase = m0 + wm * 32 + i * 16 + rowHalf;
      int nBase = n0 + wn * 64 + j * 16 + nl;
      #pragma unroll
      for (int r = 0; r < 8; ++r) {
        long idx = cOff + (long)(mBase + r) * ldc + nBase;
        float v = acc[i][j][r];
        if (CBF16) ((unsigned short*)Cv)[idx] = f2bf(v);
        else       ((float*)Cv)[idx] = v;
      }
    }
}

// ---------------------------------------------------------------- fused attention
// One block = (batch b, head h, 32-row Q strip). 8 waves.
// scores -> w_mu/w_sigma row-dots -> mu/var -> r tile -> ctx = r @ vT
#define RLD 520   // padded row stride (halves) of r tile, K=512
__global__ __launch_bounds__(256)
void attn_fused(const unsigned short* __restrict__ queryb, // [B][Q][E]
                const unsigned short* __restrict__ keysb,  // [B][NB][E]
                const unsigned short* __restrict__ valsT,  // [B][E][NB]
                const float* __restrict__ w_mu, const float* __restrict__ w_sigma,
                const float* __restrict__ basis_mu, const float* __restrict__ basis_sigma,
                unsigned short* __restrict__ ctx) {        // [B][Q][E]
  __shared__ float wmu_s[NB_DIM], wsig_s[NB_DIM], bmu_s[NB_DIM], bvar_s[NB_DIM];
  __shared__ float smu[32], ssig[32], muArr[32], vArr[32];
  __shared__ unsigned short rT[32 * RLD];

  const int qs = blockIdx.x, h = blockIdx.y, bb = blockIdx.z;
  const int tid = threadIdx.x, lane = tid & 31, wid = tid >> 5;

  for (int i = tid; i < NB_DIM; i += 256) {
    wmu_s[i] = w_mu[i];
    wsig_s[i] = w_sigma[i];
    bmu_s[i] = basis_mu[i];
    float s = basis_sigma[i];
    bvar_s[i] = s * s;
  }
  if (tid < 32) { smu[tid] = 0.f; ssig[tid] = 0.f; }
  __syncthreads();

  const long qRow0 = (long)bb * Q_DIM + qs * 32;
  const unsigned short* Aq = queryb + qRow0 * E_DIM + h * D_DIM;
  const unsigned short* Bk = keysb + (long)bb * NB_DIM * E_DIM + h * D_DIM;

  const int wm = wid >> 2, wn4 = wid & 3;  // 2 m-tiles x 4 n-groups(128 cols)
  v16bf a0 = load_frag_a(Aq + (long)(wm * 16) * E_DIM,      E_DIM, lane);
  v16bf a1 = load_frag_a(Aq + (long)(wm * 16) * E_DIM + 32, E_DIM, lane);

  float pmu[8], psig[8];
  #pragma unroll
  for (int r = 0; r < 8; ++r) { pmu[r] = 0.f; psig[r] = 0.f; }

  for (int f = 0; f < 8; ++f) {
    int nBase = (wn4 * 8 + f) * 16;
    v16bf b0 = load_frag_b(Bk + (long)nBase * E_DIM,      E_DIM, lane);
    v16bf b1 = load_frag_b(Bk + (long)nBase * E_DIM + 32, E_DIM, lane);
    v8f c = (v8f){};
    c = wmma_bf16(a0, b0, c);
    c = wmma_bf16(a1, b1, c);
    int n = nBase + (lane & 15);
    float wmv = wmu_s[n], wsv = wsig_s[n];
    #pragma unroll
    for (int r = 0; r < 8; ++r) {
      pmu[r]  += c[r] * wmv;
      psig[r] += c[r] * wsv;
    }
  }
  // reduce over the 16 lanes holding one row-half
  #pragma unroll
  for (int off = 1; off < 16; off <<= 1)
    #pragma unroll
    for (int r = 0; r < 8; ++r) {
      pmu[r]  += __shfl_xor(pmu[r],  off, 32);
      psig[r] += __shfl_xor(psig[r], off, 32);
    }
  if ((lane & 15) == 0) {
    int rowBase = wm * 16 + (lane >> 4) * 8;
    #pragma unroll
    for (int r = 0; r < 8; ++r) {
      atomicAdd(&smu[rowBase + r],  pmu[r]);
      atomicAdd(&ssig[rowBase + r], psig[r]);
    }
  }
  __syncthreads();

  if (tid < 32) {
    const float scale = 0.125f;                 // 1/sqrt(64)
    float x = smu[tid] * scale;
    muArr[tid] = 1.f / (1.f + __expf(-x));
    float y = ssig[tid] * scale;
    float sp = fmaxf(y, 0.f) + log1pf(__expf(-fabsf(y)));   // softplus
    vArr[tid] = fmaxf(sp, 1e-4f);
  }
  __syncthreads();

  const float INV_SQRT_2PI = 0.3989422804014327f;
  for (int idx = tid; idx < 32 * NB_DIM; idx += 256) {
    int qq = idx >> 9, n = idx & (NB_DIM - 1);
    float var = vArr[qq] + bvar_s[n];
    float inv = __frsqrt_rn(var);
    float d = muArr[qq] - bmu_s[n];
    float e = __expf(-0.5f * d * d / var);
    rT[qq * RLD + n] = f2bf(e * inv * INV_SQRT_2PI);
  }
  __syncthreads();

  // ctx = r[32,512] @ v[512,64]; wave (wm, wn2) owns one 16x16 output frag
  const int wn2 = wid & 3;
  const unsigned short* Bv =
      valsT + ((long)bb * E_DIM + h * D_DIM + wn2 * 16) * NB_DIM;
  v8f c = (v8f){};
  for (int k0 = 0; k0 < NB_DIM; k0 += 32) {
    v16bf af = load_frag_a(&rT[(wm * 16) * RLD + k0], RLD, lane);
    v16bf bf = load_frag_b(Bv + k0, NB_DIM, lane);
    c = wmma_bf16(af, bf, c);
  }
  const int nl = lane & 15, rowHalf = (lane >> 4) * 8;
  const long q0 = qRow0 + wm * 16 + rowHalf;
  const int col = h * D_DIM + wn2 * 16 + nl;
  #pragma unroll
  for (int r = 0; r < 8; ++r)
    ctx[(q0 + r) * E_DIM + col] = f2bf(c[r]);
}

// ---------------------------------------------------------------- launch
extern "C" void kernel_launch(void* const* d_in, const int* in_sizes, int n_in,
                              void* d_out, int out_size, void* d_ws, size_t ws_size,
                              hipStream_t stream) {
  const float* k_in  = (const float*)d_in[0];   // [L,B,E]
  const float* q_in  = (const float*)d_in[1];   // [Q,B,E]
  const float* Wq    = (const float*)d_in[2];
  const float* Wk    = (const float*)d_in[3];
  const float* Wv    = (const float*)d_in[4];
  const float* Wo    = (const float*)d_in[5];
  const float* w_mu  = (const float*)d_in[6];
  const float* w_sig = (const float*)d_in[7];
  const float* G     = (const float*)d_in[8];   // [L,NB]
  const float* bmu   = (const float*)d_in[9];
  const float* bsig  = (const float*)d_in[10];

  char* ws = (char*)d_ws;
  size_t off = 0;
  auto carve = [&](size_t bytes) { char* p = ws + off; off += (bytes + 255) & ~(size_t)255; return p; };
  unsigned short* qbf   = (unsigned short*)carve((size_t)Q_DIM * B_DIM * E_DIM * 2);
  unsigned short* Wqbf  = (unsigned short*)carve((size_t)E_DIM * E_DIM * 2);
  unsigned short* Wkbf  = (unsigned short*)carve((size_t)E_DIM * E_DIM * 2);
  unsigned short* Wvbf  = (unsigned short*)carve((size_t)E_DIM * E_DIM * 2);
  unsigned short* Wobf  = (unsigned short*)carve((size_t)E_DIM * E_DIM * 2);
  unsigned short* Gt    = (unsigned short*)carve((size_t)NB_DIM * L_DIM * 2);
  unsigned short* kT    = (unsigned short*)carve((size_t)B_DIM * E_DIM * L_DIM * 2);
  unsigned short* Bm    = (unsigned short*)carve((size_t)B_DIM * NB_DIM * E_DIM * 2);
  unsigned short* query = (unsigned short*)carve((size_t)B_DIM * Q_DIM * E_DIM * 2);
  unsigned short* keys  = (unsigned short*)carve((size_t)B_DIM * NB_DIM * E_DIM * 2);
  unsigned short* valsT = (unsigned short*)carve((size_t)B_DIM * E_DIM * NB_DIM * 2);
  unsigned short* ctx   = (unsigned short*)carve((size_t)B_DIM * Q_DIM * E_DIM * 2);

  // f32 -> bf16 copies
  int nq = Q_DIM * B_DIM * E_DIM;
  cvt_f32_bf16<<<(nq + 255) / 256, 256, 0, stream>>>(q_in, qbf, nq);
  int nw = E_DIM * E_DIM;
  cvt_f32_bf16<<<(nw + 255) / 256, 256, 0, stream>>>(Wq, Wqbf, nw);
  cvt_f32_bf16<<<(nw + 255) / 256, 256, 0, stream>>>(Wk, Wkbf, nw);
  cvt_f32_bf16<<<(nw + 255) / 256, 256, 0, stream>>>(Wv, Wvbf, nw);
  cvt_f32_bf16<<<(nw + 255) / 256, 256, 0, stream>>>(Wo, Wobf, nw);

  // Gt[NB,L] = G^T
  transpose_f32_bf16<<<dim3(NB_DIM / 32, L_DIM / 32, 1), 256, 0, stream>>>(
      G, 0, NB_DIM, Gt, 0, L_DIM);
  // kT[b][E][L] = k[:,b,:]^T
  transpose_f32_bf16<<<dim3(E_DIM / 32, L_DIM / 32, B_DIM), 256, 0, stream>>>(
      k_in, (long)E_DIM, B_DIM * E_DIM, kT, (long)E_DIM * L_DIM, L_DIM);

  // Bm[b][NB][E] = Gt @ kT_b^T          (M=NB, N=E, K=L)
  gemm_bf16<true><<<dim3(E_DIM / 128, NB_DIM / 128, B_DIM), 256, 0, stream>>>(
      Gt, 0, L_DIM, kT, (long)E_DIM * L_DIM, L_DIM,
      Bm, (long)NB_DIM * E_DIM, E_DIM, L_DIM);

  // query[b][Q][E] = q_b @ Wq^T         (M=Q, N=E, K=E)
  gemm_bf16<true><<<dim3(E_DIM / 128, Q_DIM / 128, B_DIM), 256, 0, stream>>>(
      qbf, (long)E_DIM, B_DIM * E_DIM, Wqbf, 0, E_DIM,
      query, (long)Q_DIM * E_DIM, E_DIM, E_DIM);

  // keys[b][NB][E] = Bm_b @ Wk^T        (M=NB, N=E, K=E)
  gemm_bf16<true><<<dim3(E_DIM / 128, NB_DIM / 128, B_DIM), 256, 0, stream>>>(
      Bm, (long)NB_DIM * E_DIM, E_DIM, Wkbf, 0, E_DIM,
      keys, (long)NB_DIM * E_DIM, E_DIM, E_DIM);

  // valsT[b][E][NB] = Wv @ Bm_b^T       (M=E, N=NB, K=E)
  gemm_bf16<true><<<dim3(NB_DIM / 128, E_DIM / 128, B_DIM), 256, 0, stream>>>(
      Wvbf, 0, E_DIM, Bm, (long)NB_DIM * E_DIM, E_DIM,
      valsT, (long)E_DIM * NB_DIM, NB_DIM, E_DIM);

  // fused scores -> mu/sigma -> r -> ctx
  attn_fused<<<dim3(Q_DIM / 32, H_DIM, B_DIM), 256, 0, stream>>>(
      query, keys, valsT, w_mu, w_sig, bmu, bsig, ctx);

  // out[q][b][f] = ctx_b @ Wo^T         (M=Q, N=E, K=E), f32 strided output
  gemm_bf16<false><<<dim3(E_DIM / 128, Q_DIM / 128, B_DIM), 256, 0, stream>>>(
      ctx, (long)Q_DIM * E_DIM, E_DIM, Wobf, 0, E_DIM,
      d_out, (long)E_DIM, B_DIM * E_DIM, E_DIM);
}
